// ChEst_13082470384323
// MI455X (gfx1250) — matrix operand order repaired and checked by reading, not compiled, not run on hardware
//
#include <hip/hip_runtime.h>
#include <hip/hip_bf16.h>

typedef __bf16 v16bf __attribute__((ext_vector_type(16)));
typedef float  v8f   __attribute__((ext_vector_type(8)));
typedef float  v4f   __attribute__((ext_vector_type(4)));
typedef float  v2f   __attribute__((ext_vector_type(2)));

#define COL 512
#define SYM 16384

__device__ __forceinline__ float fast_tanh(float x) {
#if __has_builtin(__builtin_amdgcn_tanhf)
    return __builtin_amdgcn_tanhf(x);   // gfx1250 v_tanh_f32
#else
    return tanhf(x);
#endif
}

// ---------------------------------------------------------------------------
// Phase 1: Z = X @ W_ih^T + (b_ih + b_hh), fp32 WMMA (16x16x4), one 16x16
// output tile per wave. 32768 tiles -> 4096 blocks x 8 waves. Bandwidth-
// trivial on MI455X (~2 GB L2-side traffic, W_ih stays hot in L2).
// ---------------------------------------------------------------------------
__global__ __launch_bounds__(256) void ChEst_phase1_gemm(
    const float* __restrict__ X, const float* __restrict__ Wih,
    const float* __restrict__ bih, const float* __restrict__ bhh,
    float* __restrict__ Z)
{
    const int lane = threadIdx.x & 31;
    const int wave = threadIdx.x >> 5;
    const int tile = blockIdx.x * 8 + wave;   // 0..32767
    const int tM   = tile >> 5;               // 0..1023 (rows of X, /16)
    const int tN   = tile & 31;               // 0..31   (rows of W_ih, /16)
    const int l16  = lane & 15;
    const int koff = (lane < 16) ? 0 : 2;

    const float* arow = X   + (size_t)(tM * 16 + l16) * COL + koff;
    const float* brow = Wih + (size_t)(tN * 16 + l16) * COL + koff;

    v8f acc = {};
#pragma unroll 8
    for (int k0 = 0; k0 < COL; k0 += 4) {
        v2f a = *(const v2f*)(arow + k0);
        v2f b = *(const v2f*)(brow + k0);
        acc = __builtin_amdgcn_wmma_f32_16x16x4_f32(
            false, a, false, b, (short)0, acc, false, false);
    }

    // C/D layout: VGPR r holds (M = r + (lane<16?0:8), N = lane%16).
    const int   n    = tN * 16 + l16;
    const float bias = bih[n] + bhh[n];
    const int   m0   = tM * 16 + ((lane < 16) ? 0 : 8);
#pragma unroll
    for (int r = 0; r < 8; ++r)
        Z[(size_t)(m0 + r) * COL + n] = acc[r] + bias;
}

// ---------------------------------------------------------------------------
// Phase 2: sequential scan h_t = tanh(Z[t] + h_{t-1} @ W_hh^T), in-place in
// d_out. Single 1024-thread workgroup (32 waves on one WGP). W_hh lives
// permanently in registers as bf16 WMMA A-fragments (128 VGPRs/thread);
// h is double-buffered in LDS as bf16 so the replicated-column B fragment is
// two contiguous ds_load_b128s per K-chunk. Z row t+1 is prefetched into
// registers before the WMMA chain of step t so HBM/L2 latency is hidden.
// Two interleaved accumulators halve the WMMA D->C dependency chain.
// ---------------------------------------------------------------------------
__global__ __launch_bounds__(1024, 1) void ChEst_phase2_scan(
    const float* __restrict__ Whh, float* __restrict__ out)
{
    __shared__ __attribute__((aligned(64))) __bf16 hbuf[2][COL];

    const int  lane = threadIdx.x & 31;
    const int  wave = threadIdx.x >> 5;       // 0..31 -> output rows 16w..16w+15
    const int  l16  = lane & 15;
    const bool lo   = lane < 16;
    const int  m    = wave * 16 + l16;        // W_hh row this lane carries
    const int  koff = lo ? 0 : 16;

    // Preload all of W_hh (each element exactly once across the workgroup)
    // into bf16 A fragments: 16 chunks of K=32 -> 128 VGPRs per thread.
    // A layout (16x32 MxK): lane L, row M=L%16; elems 0..15 -> K=k0+(L<16?0:16)+e
    // (contiguous, matching the B fragment).
    v16bf a[16];
#pragma unroll
    for (int c = 0; c < 16; ++c) {
        const float* wr = Whh + (size_t)m * COL + c * 32 + koff;
#pragma unroll
        for (int e = 0; e < 8; ++e) {
            a[c][e]     = (__bf16)wr[e];
            a[c][e + 8] = (__bf16)wr[8 + e];
        }
    }

    // h_{-1} = 0
    for (int i = threadIdx.x; i < 2 * COL; i += 1024)
        ((__bf16*)hbuf)[i] = (__bf16)0.0f;
    __syncthreads();

    const int boff = lo ? 0 : 16;             // B-frag K offset per lane half
    const int mo   = wave * 16 + (lo ? 0 : 8);// output row base for C fragment
    const float* zp = out + mo;

    // Pipeline prime: Z[0] for this lane's 8 output rows (contiguous 32B).
    v4f z0 = *(const v4f*)(zp);
    v4f z1 = *(const v4f*)(zp + 4);

    int p = 0;
    for (int t = 0; t < SYM; ++t) {
        // Prefetch Z[t+1] (clamped) -- latency hidden under the WMMA chain.
        const size_t nx = (size_t)((t + 1 < SYM) ? t + 1 : t) * COL;
        v4f n0 = *(const v4f*)(zp + nx);
        v4f n1 = *(const v4f*)(zp + nx + 4);

        // B layout (32x16 KxN, all columns = h): lane L loads 16 contiguous
        // bf16 at K = k0 + (L<16?0:16).
        v8f acc0 = {}, acc1 = {};
#pragma unroll
        for (int c = 0; c < 16; c += 2) {
            v16bf b0 = *(const v16bf*)&hbuf[p][c * 32 + boff];
            v16bf b1 = *(const v16bf*)&hbuf[p][(c + 1) * 32 + boff];
            acc0 = __builtin_amdgcn_wmma_f32_16x16x32_bf16(
                false, a[c], false, b0, (short)0, acc0, false, false);
            acc1 = __builtin_amdgcn_wmma_f32_16x16x32_bf16(
                false, a[c + 1], false, b1, (short)0, acc1, false, false);
        }

        // Every column of D is identical (= W_slab @ h); lane carries rows
        // mo..mo+7. Overwrite Z[t] with h_t in place (store data-depends on
        // the earlier load of the same address, so ordering is safe).
        float* zw = out + (size_t)t * COL + mo;
#pragma unroll
        for (int r = 0; r < 8; ++r) {
            const float z  = (r < 4) ? z0[r] : z1[r - 4];
            const float hn = fast_tanh(acc0[r] + acc1[r] + z);
            if (l16 == 0) {                   // lanes 0 and 16 commit results
                zw[r]               = hn;
                hbuf[p ^ 1][mo + r] = (__bf16)hn;
            }
        }
        z0 = n0;
        z1 = n1;
        __syncthreads();                      // publish h_t before next step
        p ^= 1;
    }
}

extern "C" void kernel_launch(void* const* d_in, const int* in_sizes, int n_in,
                              void* d_out, int out_size, void* d_ws, size_t ws_size,
                              hipStream_t stream) {
    const float* X   = (const float*)d_in[0];  // [16384, 512]
    const float* Wih = (const float*)d_in[1];  // [512, 512]
    const float* Whh = (const float*)d_in[2];  // [512, 512]
    const float* bih = (const float*)d_in[3];  // [512]
    const float* bhh = (const float*)d_in[4];  // [512]
    float* out = (float*)d_out;                // [16384, 512]

    // Phase 1: parallel input projection -> out (used as Z scratch).
    ChEst_phase1_gemm<<<4096, 256, 0, stream>>>(X, Wih, bih, bhh, out);
    // Phase 2: sequential recurrence, in-place Z -> h on a single WGP.
    ChEst_phase2_scan<<<1, 1024, 0, stream>>>(Whh, out);
}
